// MultiHeadAttentionRope_76527727280147
// MI455X (gfx1250) — matrix operand order repaired and checked by reading, not compiled, not run on hardware
//
#include <hip/hip_runtime.h>

typedef __bf16 bf16_t;
typedef __attribute__((ext_vector_type(8)))  __bf16 v8bf;
typedef __attribute__((ext_vector_type(16))) __bf16 v16bf;
typedef __attribute__((ext_vector_type(8)))  float  v8f;

namespace {
constexpr int kDModel = 1024;
constexpr int kHeads  = 16;
constexpr int kDk     = 64;
constexpr int kSeq    = 2048;
constexpr int kBatch  = 2;
constexpr int kRows   = kBatch * kSeq;   // 4096
constexpr int kNqkv   = 3 * kDModel;     // 3072
}

__device__ __forceinline__ v8f wmma_bf16(v16bf a, v16bf b, v8f c) {
  // (neg_a, A, neg_b, B, c_mod, C, reuse_a, reuse_b)
  return __builtin_amdgcn_wmma_f32_16x16x32_bf16(false, a, false, b, (short)0, c,
                                                 false, false);
}

// A-matrix fragment (16 rows x 32 K) from row-major bf16 [*, ld].
// ISA 7.12.2: lane holds row = lane%16; lanes 0-15 carry K {0..7,16..23},
// lanes 16-31 carry K {8..15,24..31}.
__device__ __forceinline__ v16bf load_a_frag(const bf16_t* base, int ld, int row,
                                             int k0, int lane) {
  int r  = row + (lane & 15);
  int kh = (lane >> 4) << 3;  // 0 or 8
  const bf16_t* p = base + (size_t)r * ld + k0 + kh;
  v8bf lo = *(const v8bf*)(p);
  v8bf hi = *(const v8bf*)(p + 16);
  return __builtin_shufflevector(lo, hi, 0, 1, 2, 3, 4, 5, 6, 7,
                                 8, 9, 10, 11, 12, 13, 14, 15);
}

// B-matrix fragment (32 K x 16 cols) sourced from B^T row-major [N, ld].
// ISA 7.12.2: lane holds col = lane%16; lanes 0-15 carry K {0..7,16..23},
// lanes 16-31 carry K {8..15,24..31}.
__device__ __forceinline__ v16bf load_b_frag(const bf16_t* base, int ld, int col,
                                             int k0, int lane) {
  int r  = col + (lane & 15);
  int kh = (lane >> 4) << 3;  // 0 or 8
  const bf16_t* p = base + (size_t)r * ld + k0 + kh;
  v8bf lo = *(const v8bf*)(p);
  v8bf hi = *(const v8bf*)(p + 16);
  return __builtin_shufflevector(lo, hi, 0, 1, 2, 3, 4, 5, 6, 7,
                                 8, 9, 10, 11, 12, 13, 14, 15);
}

__device__ __forceinline__ v8f v8f_zero() {
  v8f z;
#pragma unroll
  for (int i = 0; i < 8; ++i) z[i] = 0.0f;
  return z;
}

// ---------------------------------------------------------------------------
// fp32 -> bf16 convert (grid-stride)
// ---------------------------------------------------------------------------
__global__ void cvt_f32_to_bf16(const float* __restrict__ in,
                                bf16_t* __restrict__ out, int n) {
  int i = blockIdx.x * blockDim.x + threadIdx.x;
  int stride = gridDim.x * blockDim.x;
  for (; i < n; i += stride) out[i] = (bf16_t)in[i];
}

// ---------------------------------------------------------------------------
// C[M,N] (fp32) = A[M,K] (bf16 rowmajor) * W[N,K]^T (bf16 rowmajor)
// One wave computes a 64x64 tile: 16 wmma per 32-wide K step.
// ---------------------------------------------------------------------------
__global__ void __launch_bounds__(256)
gemm_bf16_wmma(const bf16_t* __restrict__ A, const bf16_t* __restrict__ W,
               float* __restrict__ C, int M, int N, int K) {
  int wid  = (blockIdx.x * blockDim.x + threadIdx.x) >> 5;
  int lane = threadIdx.x & 31;
  int ntiles = N >> 6;
  int tm = wid / ntiles;
  int tn = wid % ntiles;
  if (tm * 64 >= M) return;  // wave-uniform

  v8f acc[4][4];
#pragma unroll
  for (int i = 0; i < 4; ++i)
#pragma unroll
    for (int j = 0; j < 4; ++j) acc[i][j] = v8f_zero();

  for (int k0 = 0; k0 < K; k0 += 32) {
    v16bf a[4];
#pragma unroll
    for (int i = 0; i < 4; ++i)
      a[i] = load_a_frag(A, K, tm * 64 + i * 16, k0, lane);
#pragma unroll
    for (int j = 0; j < 4; ++j) {
      v16bf b = load_b_frag(W, K, tn * 64 + j * 16, k0, lane);
#pragma unroll
      for (int i = 0; i < 4; ++i) acc[i][j] = wmma_bf16(a[i], b, acc[i][j]);
    }
  }

  // C layout: lane holds col = lane%16, rows (lane/16)*8 + v
  int rbase = tm * 64 + ((lane >> 4) << 3);
  int cbase = tn * 64 + (lane & 15);
#pragma unroll
  for (int i = 0; i < 4; ++i)
#pragma unroll
    for (int j = 0; j < 4; ++j)
#pragma unroll
      for (int v = 0; v < 8; ++v)
        C[(size_t)(rbase + i * 16 + v) * N + cbase + j * 16] = acc[i][j][v];
}

// ---------------------------------------------------------------------------
// RoPE epilogue + repack. proj is [4096, 3072] fp32 (Q|K|V per 1024 cols).
// The reference tridiagonal rope matrix == pairwise rotation with angle
// pos * theta^(-2*dp/64) shared by the (2dp, 2dp+1) pair.
// Outputs: Qr,Kr bf16 [B*H, S, 64]; Vt bf16 [B*H, 64, S] (transposed for PV).
// ---------------------------------------------------------------------------
__global__ void rope_pack(const float* __restrict__ proj,
                          const int* __restrict__ pos,
                          bf16_t* __restrict__ Qr, bf16_t* __restrict__ Kr,
                          bf16_t* __restrict__ Vt) {
  int idx = blockIdx.x * blockDim.x + threadIdx.x;  // [0, B*S*H*32)
  int dp = idx & 31;
  int h  = (idx >> 5) & (kHeads - 1);
  int s  = (idx >> 9) & (kSeq - 1);
  int b  = idx >> 20;

  size_t row = (size_t)(b * kSeq + s);
  const float* pr = proj + row * kNqkv + h * kDk + dp * 2;
  float q0 = pr[0],            q1 = pr[1];
  float k0 = pr[kDModel],      k1 = pr[kDModel + 1];
  float v0 = pr[2 * kDModel],  v1 = pr[2 * kDModel + 1];

  float p   = (float)pos[s];
  float inv = __expf(-(float)dp * (2.0f / (float)kDk) * 9.210340371976184f);
  float sn, cs;
  __sincosf(p * inv, &sn, &cs);

  size_t hb = (size_t)(b * kHeads + h);
  bf16_t* q = Qr + hb * kSeq * kDk + (size_t)s * kDk + dp * 2;
  q[0] = (bf16_t)(cs * q0 - sn * q1);
  q[1] = (bf16_t)(sn * q0 + cs * q1);
  bf16_t* k = Kr + hb * kSeq * kDk + (size_t)s * kDk + dp * 2;
  k[0] = (bf16_t)(cs * k0 - sn * k1);
  k[1] = (bf16_t)(sn * k0 + cs * k1);
  bf16_t* vt = Vt + hb * (size_t)kDk * kSeq;
  vt[(size_t)(dp * 2) * kSeq + s]     = (bf16_t)v0;
  vt[(size_t)(dp * 2 + 1) * kSeq + s] = (bf16_t)v1;
}

// ---------------------------------------------------------------------------
// Flash-style causal attention. One wave owns 16 query rows of one (b,h).
// Per 64-key block: 8 wmma for S = Q K^T, fp32 online softmax, P through a
// private LDS slice (C-layout -> A-layout), 8 wmma for O += P V.
// Output written as bf16 in concat layout [B*S, H*dk].
// ---------------------------------------------------------------------------
__global__ void __launch_bounds__(256)
attn_flash(const bf16_t* __restrict__ Qr, const bf16_t* __restrict__ Kr,
           const bf16_t* __restrict__ Vt, bf16_t* __restrict__ Out) {
  __shared__ bf16_t Pt[8][16 * 64];

  int lane = threadIdx.x & 31;
  int wib  = threadIdx.x >> 5;
  int wid  = blockIdx.x * 8 + wib;
  int qt = wid & ((kSeq / 16) - 1);  // 128 query tiles
  int bh = wid >> 7;                 // [0, 32)
  int b = bh >> 4, h = bh & (kHeads - 1);

  const bf16_t* Qh = Qr + (size_t)bh * kSeq * kDk;
  const bf16_t* Kh = Kr + (size_t)bh * kSeq * kDk;
  const bf16_t* Vh = Vt + (size_t)bh * kDk * kSeq;
  int q0 = qt * 16;

  v16bf qa0 = load_a_frag(Qh, kDk, q0, 0,  lane);
  v16bf qa1 = load_a_frag(Qh, kDk, q0, 32, lane);

  v8f o[4];
  float mrun[8], lrun[8];
#pragma unroll
  for (int f = 0; f < 4; ++f) o[f] = v8f_zero();
#pragma unroll
  for (int v = 0; v < 8; ++v) { mrun[v] = -1.0e30f; lrun[v] = 0.0f; }

  const float scale = 0.125f;  // 1/sqrt(64)
  int rowbase = q0 + ((lane >> 4) << 3);

  for (int kb = 0; kb <= q0 + 15; kb += 64) {
    // scores 16x64
    v8f s[4];
#pragma unroll
    for (int f = 0; f < 4; ++f) s[f] = v8f_zero();
#pragma unroll
    for (int f = 0; f < 4; ++f) {
      v16bf kb0 = load_b_frag(Kh, kDk, kb + f * 16, 0,  lane);
      v16bf kb1 = load_b_frag(Kh, kDk, kb + f * 16, 32, lane);
      s[f] = wmma_bf16(qa0, kb0, s[f]);
      s[f] = wmma_bf16(qa1, kb1, s[f]);
    }
    // scale + causal mask + row max
    float mnew[8];
#pragma unroll
    for (int v = 0; v < 8; ++v) mnew[v] = mrun[v];
#pragma unroll
    for (int f = 0; f < 4; ++f) {
      int key = kb + f * 16 + (lane & 15);
#pragma unroll
      for (int v = 0; v < 8; ++v) {
        float val = s[f][v] * scale;
        if (key > rowbase + v) val = -1.0e30f;
        s[f][v] = val;
        mnew[v] = fmaxf(mnew[v], val);
      }
    }
#pragma unroll
    for (int off = 1; off < 16; off <<= 1)
#pragma unroll
      for (int v = 0; v < 8; ++v)
        mnew[v] = fmaxf(mnew[v], __shfl_xor(mnew[v], off, 32));
    // exp + row sum
    float psum[8];
#pragma unroll
    for (int v = 0; v < 8; ++v) psum[v] = 0.0f;
#pragma unroll
    for (int f = 0; f < 4; ++f)
#pragma unroll
      for (int v = 0; v < 8; ++v) {
        float pv = __expf(s[f][v] - mnew[v]);
        s[f][v] = pv;
        psum[v] += pv;
      }
#pragma unroll
    for (int off = 1; off < 16; off <<= 1)
#pragma unroll
      for (int v = 0; v < 8; ++v) psum[v] += __shfl_xor(psum[v], off, 32);
    // online rescale
#pragma unroll
    for (int v = 0; v < 8; ++v) {
      float corr = __expf(mrun[v] - mnew[v]);
      lrun[v] = lrun[v] * corr + psum[v];
      mrun[v] = mnew[v];
#pragma unroll
      for (int f = 0; f < 4; ++f) o[f][v] *= corr;
    }
    // P: C-layout -> LDS (row-major 16x64) -> A-layout. Same-wave LDS is
    // in-order (DScnt), private slice => no barrier needed.
    bf16_t* Pl = &Pt[wib][0];
#pragma unroll
    for (int f = 0; f < 4; ++f)
#pragma unroll
      for (int v = 0; v < 8; ++v)
        Pl[(((lane >> 4) << 3) + v) * 64 + f * 16 + (lane & 15)] =
            (bf16_t)s[f][v];
    v16bf pa0 = load_a_frag(Pl, 64, 0, 0,  lane);
    v16bf pa1 = load_a_frag(Pl, 64, 0, 32, lane);
    // O += P * V   (Vt rows are dk dims, contiguous over keys)
#pragma unroll
    for (int f = 0; f < 4; ++f) {
      v16bf vb0 = load_b_frag(Vh, kSeq, f * 16, kb,      lane);
      v16bf vb1 = load_b_frag(Vh, kSeq, f * 16, kb + 32, lane);
      o[f] = wmma_bf16(pa0, vb0, o[f]);
      o[f] = wmma_bf16(pa1, vb1, o[f]);
    }
  }

  size_t orow0 = (size_t)(b * kSeq + q0);
#pragma unroll
  for (int f = 0; f < 4; ++f)
#pragma unroll
    for (int v = 0; v < 8; ++v) {
      int r = ((lane >> 4) << 3) + v;
      Out[(orow0 + r) * kDModel + h * kDk + f * 16 + (lane & 15)] =
          (bf16_t)(o[f][v] / lrun[v]);
    }
}

// ---------------------------------------------------------------------------
extern "C" void kernel_launch(void* const* d_in, const int* in_sizes, int n_in,
                              void* d_out, int out_size, void* d_ws,
                              size_t ws_size, hipStream_t stream) {
  (void)in_sizes; (void)n_in; (void)out_size; (void)ws_size;
  const float* x     = (const float*)d_in[0];
  const float* Wattn = (const float*)d_in[1];
  const float* Wout  = (const float*)d_in[2];
  // d_in[3] = rope matrices (recomputed analytically on device)
  const int*   pos   = (const int*)d_in[4];
  float* out = (float*)d_out;

  char* ws = (char*)d_ws;
  size_t off = 0;
  auto alloc = [&](size_t bytes) -> char* {
    char* p = ws + off;
    off += (bytes + 255) & ~(size_t)255;
    return p;
  };
  bf16_t* xb   = (bf16_t*)alloc((size_t)kRows * kDModel * 2);   // 8 MiB
  bf16_t* wab  = (bf16_t*)alloc((size_t)kNqkv * kDModel * 2);   // 6 MiB
  bf16_t* wob  = (bf16_t*)alloc((size_t)kDModel * kDModel * 2); // 2 MiB
  float*  proj = (float*)alloc((size_t)kRows * kNqkv * 4);      // 48 MiB
  bf16_t* Qr   = (bf16_t*)alloc((size_t)kBatch * kHeads * kSeq * kDk * 2);
  bf16_t* Kr   = (bf16_t*)alloc((size_t)kBatch * kHeads * kSeq * kDk * 2);
  bf16_t* Vt   = (bf16_t*)alloc((size_t)kBatch * kHeads * kSeq * kDk * 2);
  bf16_t* att  = (bf16_t*)alloc((size_t)kRows * kDModel * 2);   // 8 MiB

  cvt_f32_to_bf16<<<1024, 256, 0, stream>>>(x, xb, kRows * kDModel);
  cvt_f32_to_bf16<<<1024, 256, 0, stream>>>(Wattn, wab, kNqkv * kDModel);
  cvt_f32_to_bf16<<<1024, 256, 0, stream>>>(Wout, wob, kDModel * kDModel);

  {
    int waves = (kRows / 64) * (kNqkv / 64);  // 3072
    gemm_bf16_wmma<<<waves * 32 / 256, 256, 0, stream>>>(xb, wab, proj, kRows,
                                                         kNqkv, kDModel);
  }
  {
    int total = kBatch * kSeq * kHeads * (kDk / 2);  // 2,097,152
    rope_pack<<<total / 256, 256, 0, stream>>>(proj, pos, Qr, Kr, Vt);
  }
  {
    int waves = kBatch * kHeads * (kSeq / 16);  // 4096
    attn_flash<<<waves * 32 / 256, 256, 0, stream>>>(Qr, Kr, Vt, att);
  }
  {
    int waves = (kRows / 64) * (kDModel / 64);  // 1024
    gemm_bf16_wmma<<<waves * 32 / 256, 256, 0, stream>>>(att, wob, out, kRows,
                                                         kDModel, kDModel);
  }
}
